// EncoderBlock_41910290874751
// MI455X (gfx1250) — matrix operand order repaired and checked by reading, not compiled, not run on hardware
//
#include <hip/hip_runtime.h>
#include <hip/hip_bf16.h>
#include <math.h>

#define D_MODEL 1024
#define NHEAD   16
#define DHEAD   64
#define FF_DIM  4096
#define SEQ     2048
#define BATCH   2
#define ROWS    (BATCH*SEQ)   /* 4096 token rows */
#define LN_EPS  1e-6f

typedef __bf16 bf16_t;
typedef bf16_t v16bf __attribute__((ext_vector_type(16)));
typedef float  v8f   __attribute__((ext_vector_type(8)));

union FragAB { unsigned int u[8]; v16bf v; };

// ---- fp32 -> bf16 round-to-nearest-even ------------------------------------
__device__ __forceinline__ unsigned short f2bf(float f) {
    union { float f; unsigned int u; } c; c.f = f;
    unsigned int u = c.u;
    unsigned int r = u + 0x7fffu + ((u >> 16) & 1u);
    if ((u & 0x7f800000u) == 0x7f800000u) r = u;   // Inf/NaN: truncate
    return (unsigned short)(r >> 16);
}
__device__ __forceinline__ unsigned int pack2bf(float lo, float hi) {
    return (unsigned int)f2bf(lo) | ((unsigned int)f2bf(hi) << 16);
}

// ---- CDNA5 async copy: global -> LDS, 16B per lane, ASYNCcnt tracked -------
__device__ __forceinline__ void async_copy_b128(unsigned lds_off, const void* gptr) {
    unsigned long long ga = (unsigned long long)(uintptr_t)gptr;
    asm volatile("global_load_async_to_lds_b128 %0, %1, off"
                 :: "v"(lds_off), "v"(ga) : "memory");
}

// ---- WMMA fragment loaders (bf16, 16x16x32) --------------------------------
// A (16x32, MxK) per ISA 7.12.2: lanes 0-15 hold rows M=lane with K=0..7 (V0-3)
// and K=16..23 (V4-7); lanes 16-31 hold K=8..15 / K=24..31 of the same rows.
__device__ __forceinline__ v16bf load_frag_a(const unsigned short* A, int row,
                                             int lda, int k0) {
    int lane = threadIdx.x & 31;
    int half = lane >> 4;
    FragAB f;
    const unsigned short* p = A + (size_t)row * lda + k0;
#pragma unroll
    for (int i = 0; i < 8; ++i) {
        int kb = (i < 4) ? (2*i + 8*half) : (16 + 2*(i-4) + 8*half);
        f.u[i] = *reinterpret_cast<const unsigned int*>(p + kb);
    }
    return f.v;
}
// Same A layout, but sourced from an LDS tile with padded leading dimension.
__device__ __forceinline__ v16bf lds_frag_a(const unsigned short* base, int row,
                                            int ld) {
    int lane = threadIdx.x & 31;
    int half = lane >> 4;
    FragAB f;
    const unsigned short* p = base + row * ld;
#pragma unroll
    for (int i = 0; i < 8; ++i) {
        int kb = (i < 4) ? (2*i + 8*half) : (16 + 2*(i-4) + 8*half);
        f.u[i] = *reinterpret_cast<const unsigned int*>(p + kb);
    }
    return f.v;
}
// B (32x16, KxN): same K order as A: lanes 0-15 hold K=0..7 / K=16..23 of
// column N=lane&15, lanes 16-31 hold K=8..15 / K=24..31.
// BT is the row-major [N,K] view; caller passes this lane's column index.
__device__ __forceinline__ v16bf load_frag_b(const unsigned short* BT, int col,
                                             int ldb, int k0) {
    int lane = threadIdx.x & 31;
    int half = lane >> 4;
    FragAB f;
    const unsigned short* p = BT + (size_t)col * ldb + k0;
#pragma unroll
    for (int i = 0; i < 8; ++i) {
        int kb = (i < 4) ? (2*i + 8*half) : (16 + 2*(i-4) + 8*half);
        f.u[i] = *reinterpret_cast<const unsigned int*>(p + kb);
    }
    return f.v;
}

__device__ __forceinline__ v8f wmma_bf16(v16bf a, v16bf b, v8f c) {
    return __builtin_amdgcn_wmma_f32_16x16x32_bf16(false, a, false, b,
                                                   (short)0, c, false, false);
}

// ---- weight convert + transpose: W[K,N] f32 -> WT[N,K] bf16 ----------------
__global__ __launch_bounds__(256)
void k_convert_transpose(const float* __restrict__ W,
                         unsigned short* __restrict__ WT, int K, int N) {
    int idx = blockIdx.x * 256 + threadIdx.x;
    if (idx >= K * N) return;
    int k = idx / N, n = idx % N;
    WT[(size_t)n * K + k] = f2bf(W[idx]);
}

// ---- LayerNorm (ddof=1, gamma*(x-mean)/(std+eps)+beta) -> bf16 -------------
__global__ __launch_bounds__(256)
void k_layernorm_bf16(const float* __restrict__ x,
                      const float* __restrict__ gamma,
                      const float* __restrict__ beta,
                      unsigned short* __restrict__ out, int Dn) {
    int row = blockIdx.x;
    const float* xr = x + (size_t)row * Dn;
    __shared__ float red[256];

    float s = 0.f;
    for (int i = threadIdx.x; i < Dn; i += 256) s += xr[i];
    red[threadIdx.x] = s; __syncthreads();
    for (int off = 128; off > 0; off >>= 1) {
        if ((int)threadIdx.x < off) red[threadIdx.x] += red[threadIdx.x + off];
        __syncthreads();
    }
    float mean = red[0] / Dn;
    __syncthreads();

    float vs = 0.f;
    for (int i = threadIdx.x; i < Dn; i += 256) {
        float d = xr[i] - mean; vs += d * d;
    }
    red[threadIdx.x] = vs; __syncthreads();
    for (int off = 128; off > 0; off >>= 1) {
        if ((int)threadIdx.x < off) red[threadIdx.x] += red[threadIdx.x + off];
        __syncthreads();
    }
    float stdv = sqrtf(red[0] / (float)(Dn - 1));
    float inv  = 1.f / (stdv + LN_EPS);
    for (int i = threadIdx.x; i < Dn; i += 256) {
        float v = gamma[i] * (xr[i] - mean) * inv + beta[i];
        out[(size_t)row * Dn + i] = f2bf(v);
    }
}

// ---- block-tiled WMMA GEMM: C[M,N] = A[M,K](bf16) @ BT[N,K]^T(bf16) --------
// Workgroup (8 waves, 2x4) computes a 64x256 C tile. The 64x32 A k-slab is
// double-buffered in LDS via global_load_async_to_lds_b128 (ASYNCcnt), so the
// next slab's DMA overlaps the current slab's 8 WMMAs per wave. B fragments
// stream from global (L2-resident: full B is at most 8MB vs 192MB L2).
// OUTM: 0 = row-major [M,N]; 1 = [B,H,S,DH] (Q/K); 2 = [B,H,DH,S] (V^T).
#define ATILE_LD 40   /* padded row length (bf16 elems): 80B, 16B aligned, no bank conflicts */
template <bool RELU, bool BIAS, bool RESID, int OUTM>
__global__ __launch_bounds__(256)
void k_gemm_wmma(const unsigned short* __restrict__ A,
                 const unsigned short* __restrict__ BT,
                 const float* __restrict__ bias,
                 const float* __restrict__ resid,
                 float* __restrict__ outF,
                 unsigned short* __restrict__ outB,
                 int M, int N, int K) {
    __shared__ unsigned short aTile[2][64 * ATILE_LD];

    const int tid   = threadIdx.x;
    const int lane  = tid & 31;
    const int wid   = tid >> 5;        // 0..7
    const int waveM = wid >> 2;        // 0..1  -> 32-row strip
    const int waveN = wid & 3;         // 0..3  -> 64-col strip
    const int half  = lane >> 4, r = lane & 15;

    const int nBlks = N >> 8;
    const int mBase = (blockIdx.x / nBlks) * 64;
    const int nBase = (blockIdx.x % nBlks) * 256 + waveN * 64;

    // staging map: 256 threads x 16B cover the 64x32 bf16 slab exactly
    const int sRow = tid >> 2;
    const int sCol = (tid & 3) * 8;

    v8f acc[2][4] = {};

    // prime buffer 0
    {
        unsigned loff = (unsigned)(uintptr_t)(const void*)&aTile[0][sRow * ATILE_LD + sCol];
        async_copy_b128(loff, A + (size_t)(mBase + sRow) * K + 0 + sCol);
    }

    for (int k0 = 0; k0 < K; k0 += 32) {
        const int  cur = (k0 >> 5) & 1;
        const bool pre = (k0 + 32) < K;
        if (pre) {
            unsigned loff = (unsigned)(uintptr_t)(const void*)
                            &aTile[cur ^ 1][sRow * ATILE_LD + sCol];
            async_copy_b128(loff, A + (size_t)(mBase + sRow) * K + (k0 + 32) + sCol);
        }
        if (pre) asm volatile("s_wait_asynccnt 0x1" ::: "memory");
        else     asm volatile("s_wait_asynccnt 0x0" ::: "memory");
        __syncthreads();

        v16bf a0 = lds_frag_a(&aTile[cur][0], waveM * 32 + r,      ATILE_LD);
        v16bf a1 = lds_frag_a(&aTile[cur][0], waveM * 32 + 16 + r, ATILE_LD);
#pragma unroll
        for (int t = 0; t < 4; ++t) {
            v16bf b = load_frag_b(BT, nBase + t * 16 + r, K, k0);
            acc[0][t] = wmma_bf16(a0, b, acc[0][t]);
            acc[1][t] = wmma_bf16(a1, b, acc[1][t]);
        }
        __syncthreads();   // all waves done reading before next slab overwrite
    }

#pragma unroll
    for (int mi = 0; mi < 2; ++mi) {
#pragma unroll
        for (int t = 0; t < 4; ++t) {
            int n = nBase + t * 16 + r;          // C layout: col = lane&15
#pragma unroll
            for (int vi = 0; vi < 8; ++vi) {
                int m = mBase + waveM * 32 + mi * 16 + half * 8 + vi;
                float v = acc[mi][t][vi];
                if (BIAS)  v += bias[n];
                if (RELU)  v  = fmaxf(v, 0.f);
                if (RESID) v += resid[(size_t)m * N + n];
                if (outF)  outF[(size_t)m * N + n] = v;
                if (outB) {
                    size_t idx;
                    if (OUTM == 0) {
                        idx = (size_t)m * N + n;
                    } else {
                        int b_ = m / SEQ, s_ = m % SEQ;
                        int h_ = n / DHEAD, d_ = n % DHEAD;
                        if (OUTM == 1)
                            idx = (((size_t)(b_ * NHEAD + h_) * SEQ) + s_) * DHEAD + d_;
                        else
                            idx = (((size_t)(b_ * NHEAD + h_) * DHEAD) + d_) * SEQ + s_;
                    }
                    outB[idx] = f2bf(v);
                }
            }
        }
    }
}

// ---- flash attention: 1 wave per 16-row q tile per (b,h) -------------------
// Q,K: bf16 [BH,S,DH]; Vt: bf16 [BH,DH,S]; out: bf16 [ROWS, D_MODEL]
__global__ __launch_bounds__(32)
void k_attention_flash(const unsigned short* __restrict__ Q,
                       const unsigned short* __restrict__ Kb,
                       const unsigned short* __restrict__ Vt,
                       unsigned short* __restrict__ attn_out) {
    int tile  = blockIdx.x;                 // 0 .. BH*(SEQ/16)-1
    int bh    = tile / (SEQ / 16);
    int qtile = tile % (SEQ / 16);
    int b = bh / NHEAD, h = bh % NHEAD;
    int lane = threadIdx.x & 31;
    int half = lane >> 4, r = lane & 15;

    const unsigned short* Qp = Q  + (size_t)bh * SEQ * DHEAD;
    const unsigned short* Kp = Kb + (size_t)bh * SEQ * DHEAD;
    const unsigned short* Vp = Vt + (size_t)bh * DHEAD * SEQ;

    // preload the 16xDH Q tile as two K=32 A fragments
    v16bf qa0 = load_frag_a(Qp, qtile * 16 + r, DHEAD, 0);
    v16bf qa1 = load_frag_a(Qp, qtile * 16 + r, DHEAD, 32);

    v8f o[4] = {};
    float mrow[8], lrow[8];
#pragma unroll
    for (int i = 0; i < 8; ++i) { mrow[i] = -1e30f; lrow[i] = 0.f; }

    __shared__ float p_lds[16][36];         // padded 16x32 P tile
    const float scale = 0.125f;             // 1/sqrt(64)

    for (int j0 = 0; j0 < SEQ; j0 += 32) {
        // S = Q @ K^T for 32 keys: two 16x16 C tiles, each from 2 K=32 WMMAs
        v8f s0 = {}, s1 = {};
        {
            v16bf b0 = load_frag_b(Kp, j0 + r,      DHEAD, 0);
            s0 = wmma_bf16(qa0, b0, s0);
            v16bf b1 = load_frag_b(Kp, j0 + r,      DHEAD, 32);
            s0 = wmma_bf16(qa1, b1, s0);
            v16bf b2 = load_frag_b(Kp, j0 + 16 + r, DHEAD, 0);
            s1 = wmma_bf16(qa0, b2, s1);
            v16bf b3 = load_frag_b(Kp, j0 + 16 + r, DHEAD, 32);
            s1 = wmma_bf16(qa1, b3, s1);
        }

        // online softmax update (rows live in C layout: row = half*8+vi)
        float mnew[8], alpha[8];
#pragma unroll
        for (int vi = 0; vi < 8; ++vi) {
            float v0 = s0[vi] * scale, v1 = s1[vi] * scale;
            s0[vi] = v0; s1[vi] = v1;
            float mx = fmaxf(v0, v1);
#pragma unroll
            for (int mm = 8; mm >= 1; mm >>= 1)
                mx = fmaxf(mx, __shfl_xor(mx, mm, 32));
            mnew[vi]  = fmaxf(mrow[vi], mx);
            alpha[vi] = __expf(mrow[vi] - mnew[vi]);
        }
#pragma unroll
        for (int vi = 0; vi < 8; ++vi) {
            float p0 = __expf(s0[vi] - mnew[vi]);
            float p1 = __expf(s1[vi] - mnew[vi]);
            float rs = p0 + p1;
#pragma unroll
            for (int mm = 8; mm >= 1; mm >>= 1)
                rs += __shfl_xor(rs, mm, 32);
            lrow[vi] = lrow[vi] * alpha[vi] + rs;
            mrow[vi] = mnew[vi];
            int mlocal = half * 8 + vi;
            p_lds[mlocal][r]      = p0;
            p_lds[mlocal][16 + r] = p1;
#pragma unroll
            for (int t = 0; t < 4; ++t) o[t][vi] *= alpha[vi];
        }
        __syncthreads();

        // relayout P (C layout -> A layout) through LDS, convert to bf16
        FragAB pf;
#pragma unroll
        for (int i = 0; i < 8; ++i) {
            int kb = (i < 4) ? (2*i + 8*half) : (16 + 2*(i-4) + 8*half);
            pf.u[i] = pack2bf(p_lds[r][kb], p_lds[r][kb + 1]);
        }
        __syncthreads();

        // O += P(16x32) @ V(32xDH) : Vt gives contiguous key runs per d-col
#pragma unroll
        for (int t = 0; t < 4; ++t) {
            v16bf vb = load_frag_b(Vp, t * 16 + r, SEQ, j0);
            o[t] = wmma_bf16(pf.v, vb, o[t]);
        }
    }

    // finalize: divide by row sums, scatter bf16 into [ROWS, D_MODEL]
#pragma unroll
    for (int vi = 0; vi < 8; ++vi) {
        float invl = 1.f / lrow[vi];
        int srow = qtile * 16 + half * 8 + vi;
        size_t row = (size_t)b * SEQ + srow;
#pragma unroll
        for (int t = 0; t < 4; ++t) {
            int col = h * DHEAD + t * 16 + r;
            attn_out[row * D_MODEL + col] = f2bf(o[t][vi] * invl);
        }
    }
}

// ---------------------------------------------------------------------------
extern "C" void kernel_launch(void* const* d_in, const int* in_sizes, int n_in,
                              void* d_out, int out_size, void* d_ws, size_t ws_size,
                              hipStream_t stream) {
    const float* x   = (const float*)d_in[0];
    const float* w_q = (const float*)d_in[1];
    const float* w_k = (const float*)d_in[2];
    const float* w_v = (const float*)d_in[3];
    const float* w_o = (const float*)d_in[4];
    const float* w1  = (const float*)d_in[5];
    const float* b1  = (const float*)d_in[6];
    const float* w2  = (const float*)d_in[7];
    const float* b2  = (const float*)d_in[8];
    const float* g1  = (const float*)d_in[9];
    const float* be1 = (const float*)d_in[10];
    const float* g2  = (const float*)d_in[11];
    const float* be2 = (const float*)d_in[12];
    float* out = (float*)d_out;

    char* ws = (char*)d_ws;
    const size_t MB = 1ull << 20;
    unsigned short* wqT  = (unsigned short*)(ws + 0 * MB);    // [D,D]   2MB
    unsigned short* wkT  = (unsigned short*)(ws + 2 * MB);
    unsigned short* wvT  = (unsigned short*)(ws + 4 * MB);
    unsigned short* woT  = (unsigned short*)(ws + 6 * MB);
    unsigned short* w1T  = (unsigned short*)(ws + 8 * MB);    // [FF,D]  8MB
    unsigned short* w2T  = (unsigned short*)(ws + 16 * MB);   // [D,FF]  8MB
    unsigned short* h1b  = (unsigned short*)(ws + 24 * MB);   // [ROWS,D] bf16
    unsigned short* qb   = (unsigned short*)(ws + 32 * MB);   // [B,H,S,DH]
    unsigned short* kb   = (unsigned short*)(ws + 40 * MB);   // [B,H,S,DH]
    unsigned short* vtb  = (unsigned short*)(ws + 48 * MB);   // [B,H,DH,S]
    unsigned short* attb = (unsigned short*)(ws + 56 * MB);   // [ROWS,D] bf16
    float*          x1   = (float*)        (ws + 64 * MB);    // [ROWS,D] f32
    unsigned short* h2b  = (unsigned short*)(ws + 80 * MB);   // [ROWS,D] bf16
    unsigned short* ffb  = (unsigned short*)(ws + 88 * MB);   // [ROWS,FF] bf16
    (void)ws_size; (void)in_sizes; (void)n_in; (void)out_size;

    // 1) weight convert/transpose to bf16 [N,K]
    k_convert_transpose<<<(D_MODEL*D_MODEL + 255)/256, 256, 0, stream>>>(w_q, wqT, D_MODEL, D_MODEL);
    k_convert_transpose<<<(D_MODEL*D_MODEL + 255)/256, 256, 0, stream>>>(w_k, wkT, D_MODEL, D_MODEL);
    k_convert_transpose<<<(D_MODEL*D_MODEL + 255)/256, 256, 0, stream>>>(w_v, wvT, D_MODEL, D_MODEL);
    k_convert_transpose<<<(D_MODEL*D_MODEL + 255)/256, 256, 0, stream>>>(w_o, woT, D_MODEL, D_MODEL);
    k_convert_transpose<<<(D_MODEL*FF_DIM  + 255)/256, 256, 0, stream>>>(w1,  w1T, D_MODEL, FF_DIM);
    k_convert_transpose<<<(FF_DIM*D_MODEL  + 255)/256, 256, 0, stream>>>(w2,  w2T, FF_DIM, D_MODEL);

    // 2) LN1
    k_layernorm_bf16<<<ROWS, 256, 0, stream>>>(x, g1, be1, h1b, D_MODEL);

    // 3) Q/K/V projections: grid = (M/64)*(N/256) workgroups of 8 waves
    auto gblocks = [](int M, int N) { return (M / 64) * (N / 256); };
    k_gemm_wmma<false,false,false,1><<<gblocks(ROWS, D_MODEL), 256, 0, stream>>>(
        h1b, wqT, nullptr, nullptr, nullptr, qb, ROWS, D_MODEL, D_MODEL);
    k_gemm_wmma<false,false,false,1><<<gblocks(ROWS, D_MODEL), 256, 0, stream>>>(
        h1b, wkT, nullptr, nullptr, nullptr, kb, ROWS, D_MODEL, D_MODEL);
    k_gemm_wmma<false,false,false,2><<<gblocks(ROWS, D_MODEL), 256, 0, stream>>>(
        h1b, wvT, nullptr, nullptr, nullptr, vtb, ROWS, D_MODEL, D_MODEL);

    // 4) flash attention: one wave per (b,h,16-row q tile)
    k_attention_flash<<<BATCH * NHEAD * (SEQ / 16), 32, 0, stream>>>(qb, kb, vtb, attb);

    // 5) O projection + residual -> x1
    k_gemm_wmma<false,false,true,0><<<gblocks(ROWS, D_MODEL), 256, 0, stream>>>(
        attb, woT, nullptr, x, x1, nullptr, ROWS, D_MODEL, D_MODEL);

    // 6) LN2
    k_layernorm_bf16<<<ROWS, 256, 0, stream>>>(x1, g2, be2, h2b, D_MODEL);

    // 7) FFN1: relu(h2 @ w1 + b1) -> bf16
    k_gemm_wmma<true,true,false,0><<<gblocks(ROWS, FF_DIM), 256, 0, stream>>>(
        h2b, w1T, b1, nullptr, nullptr, ffb, ROWS, FF_DIM, D_MODEL);

    // 8) FFN2: ff @ w2 + b2 + x1 -> out (f32)
    k_gemm_wmma<false,true,true,0><<<gblocks(ROWS, D_MODEL), 256, 0, stream>>>(
        ffb, w2T, b2, x1, out, nullptr, ROWS, D_MODEL, FF_DIM);
}
